// AttentionModule_34248069218321
// MI455X (gfx1250) — matrix operand order repaired and checked by reading, not compiled, not run on hardware
//
#include <hip/hip_runtime.h>
#include <hip/hip_bf16.h>
#include <math.h>

typedef __attribute__((ext_vector_type(16))) __bf16 v16bf;
typedef __attribute__((ext_vector_type(8)))  float  v8f;

// ---------------------------------------------------------------------------
// WMMA 16-bit A-matrix (16x32) per-lane K index (ISA 7.12.2):
// slot j = 2*vgpr + half  ->  K = ((j<8)? j : j+8) + 8*(lane>>4)
// ---------------------------------------------------------------------------
__device__ __forceinline__ int ka_index(int lane, int j) {
    return ((j < 8) ? j : j + 8) + 8 * (lane >> 4);
}

// Exact adaptive avg pool (mean over s x s block) of x[B=4][C=128][64][64]
__device__ __forceinline__ float pooled(const float* __restrict__ x,
                                        int b, int c, int n, int s, int Ws) {
    int hs = n / Ws, ws = n - hs * Ws;
    const float* p = x + (((size_t)b * 128 + c) * 64 + (size_t)hs * s) * 64 + (size_t)ws * s;
    float acc = 0.f;
    for (int i = 0; i < s; ++i)
        for (int j = 0; j < s; ++j)
            acc += p[i * 64 + j];
    return acc * (1.0f / (s * s));
}

// ---------------------------------------------------------------------------
// Packing kernels: pool + convert to bf16, laid out lane-major so the flash
// kernel reads each WMMA operand as one contiguous v16bf per lane.
// qa: [b][qtile][chunk(4 of 32c)][lane 32][slot 16]  (A-layout, M=row,K=c)
// kb: [b][ktile][chunk(4 of 32c)][lane 32][slot 16]  (B-layout, K=c,N=key)
// vb: [b][kchunk(N/32)][ctile 8 ][lane 32][slot 16]  (B-layout, K=key,N=c)
// ---------------------------------------------------------------------------
__global__ void pack_qa(const float* __restrict__ x, __bf16* __restrict__ qa,
                        int N, int s, int Ws) {
    int e = blockIdx.x * blockDim.x + threadIdx.x;
    if (e >= 4 * N * 128) return;
    int j = e & 15, lane = (e >> 4) & 31, chunk = (e >> 9) & 3;
    int NT = N >> 4;
    int qt = (e >> 11) % NT, b = (e >> 11) / NT;
    int c = chunk * 32 + 16 * (lane >> 4) + j;
    int n = qt * 16 + (lane & 15);
    qa[e] = (__bf16)pooled(x, b, c, n, s, Ws);
}

__global__ void pack_kb(const float* __restrict__ x, __bf16* __restrict__ kb,
                        int N, int s, int Ws) {
    int e = blockIdx.x * blockDim.x + threadIdx.x;
    if (e >= 4 * N * 128) return;
    int j = e & 15, lane = (e >> 4) & 31, chunk = (e >> 9) & 3;
    int NT = N >> 4;
    int kt = (e >> 11) % NT, b = (e >> 11) / NT;
    int c = chunk * 32 + 16 * (lane >> 4) + j;   // B-layout: K index = c
    int key = kt * 16 + (lane & 15);             // N index = key
    kb[e] = (__bf16)pooled(x, b, c, key, s, Ws);
}

__global__ void pack_vb(const float* __restrict__ x, __bf16* __restrict__ vb,
                        int N, int s, int Ws) {
    int e = blockIdx.x * blockDim.x + threadIdx.x;
    if (e >= 4 * N * 128) return;
    int j = e & 15, lane = (e >> 4) & 31, ct = (e >> 9) & 7;
    int KC = N >> 5;
    int kc = (e >> 12) % KC, b = (e >> 12) / KC;
    int c = ct * 16 + (lane & 15);               // N index = c
    int key = kc * 32 + 16 * (lane >> 4) + j;    // K index = key
    vb[e] = (__bf16)pooled(x, b, c, key, s, Ws);
}

// ---------------------------------------------------------------------------
// CDNA5 async copy: global -> LDS, 16B per lane, tracked by ASYNCcnt.
// LDS operand = byte offset (low 32 bits of the generic LDS address).
// ---------------------------------------------------------------------------
__device__ __forceinline__ void async_ld_b128(void* lds, const void* gptr) {
    unsigned loff = (unsigned)(uintptr_t)lds;
    unsigned long long g = (unsigned long long)(uintptr_t)gptr;
    asm volatile("global_load_async_to_lds_b128 %0, %1, off"
                 :: "v"(loff), "v"(g) : "memory");
}
__device__ __forceinline__ void wait_async0() {
    asm volatile("s_wait_asynccnt 0" ::: "memory");
}

// ---------------------------------------------------------------------------
// Flash attention: 1 wave = 16 query rows x full C=128 output; 4 waves/WG
// share K/V chunks staged in LDS by async-to-LDS copies, double-buffered
// one chunk ahead. S = Q K^T via 4 chained v_wmma_f32_16x16x32_bf16,
// online softmax per row through LDS, P@V via 8 WMMAs per 32-key chunk.
// out layout: [b][c][n] f32 (== [B][C][H][W] for scale 1).
// ---------------------------------------------------------------------------
__global__ __launch_bounds__(128) void flash_attn(
    const __bf16* __restrict__ qa, const __bf16* __restrict__ kb,
    const __bf16* __restrict__ vb, float* __restrict__ out, int N) {
    __shared__ __align__(16) char ldsK[2][8192];  // 2 ktiles x 4 chunks x 1KB
    __shared__ __align__(16) char ldsV[2][8192];  // 8 ctiles x 1KB
    __shared__ float sS[4][16][33];               // per-wave 16x32 score stage
    __shared__ float sRow[4][16];                 // per-wave row broadcast

    const int tid  = threadIdx.x;
    const int lane = tid & 31;
    const int w    = tid >> 5;
    const int g    = lane >> 4;
    const int row  = lane & 15;
    const int NT   = N >> 4;
    const int KC   = N >> 5;
    const int qt   = blockIdx.x * 4 + w;
    const int b    = blockIdx.y;

    const v16bf* qp = (const v16bf*)qa + ((size_t)b * NT + qt) * 4 * 32;
    const char* kg  = (const char*)kb + (size_t)b * N * 128 * 2;  // 8KB/chunk
    const char* vg  = (const char*)vb + (size_t)b * N * 128 * 2;  // 8KB/chunk

    v16bf q[4];
#pragma unroll
    for (int ch = 0; ch < 4; ++ch) q[ch] = qp[ch * 32 + lane];

    v8f o[8] = {};
    float mrow = -INFINITY, lrow = 0.f;

    // prologue: stream chunk 0 (each thread moves 4 x 16B per array)
#pragma unroll
    for (int i = 0; i < 4; ++i) {
        int off = tid * 16 + i * 2048;
        async_ld_b128(&ldsK[0][off], kg + off);
        async_ld_b128(&ldsV[0][off], vg + off);
    }

    for (int kc = 0; kc < KC; ++kc) {
        const int cur = kc & 1;
        wait_async0();       // our slice of chunk kc landed
        __syncthreads();     // everyone's slice landed
        if (kc + 1 < KC) {   // issue chunk kc+1 into the other buffer
#pragma unroll
            for (int i = 0; i < 4; ++i) {
                int off = tid * 16 + i * 2048;
                async_ld_b128(&ldsK[cur ^ 1][off], kg + (size_t)(kc + 1) * 8192 + off);
                async_ld_b128(&ldsV[cur ^ 1][off], vg + (size_t)(kc + 1) * 8192 + off);
            }
        }
        const v16bf* kl = (const v16bf*)ldsK[cur];
        const v16bf* vl = (const v16bf*)ldsV[cur];

        // ---- S tile: 16 q-rows x 32 keys ----
#pragma unroll
        for (int t = 0; t < 2; ++t) {
            v8f acc = {};
#pragma unroll
            for (int ch = 0; ch < 4; ++ch) {
                v16bf kreg = kl[(t * 4 + ch) * 32 + lane];
                acc = __builtin_amdgcn_wmma_f32_16x16x32_bf16(
                    false, q[ch], false, kreg, (short)0, acc, false, false);
            }
#pragma unroll
            for (int r = 0; r < 8; ++r)
                sS[w][r + 8 * g][t * 16 + row] = acc[r];
        }
        __syncthreads();

        // ---- online softmax; each lane owns its row (halves duplicate) ----
        float sr[32];
        float mnew = mrow;
#pragma unroll
        for (int k = 0; k < 32; ++k) {
            sr[k] = sS[w][row][k];
            mnew  = fmaxf(mnew, sr[k]);
        }
        float corr = __expf(mrow - mnew);
        float sum = 0.f;
#pragma unroll
        for (int k = 0; k < 32; ++k) {
            sr[k] = __expf(sr[k] - mnew);
            sum += sr[k];
        }
        lrow = lrow * corr + sum;
        mrow = mnew;
        if (lane < 16) sRow[w][row] = corr;
        __syncthreads();

        // rescale accumulators by this iteration's correction factor
#pragma unroll
        for (int r = 0; r < 8; ++r) {
            float cr = sRow[w][r + 8 * g];
#pragma unroll
            for (int ct = 0; ct < 8; ++ct) o[ct][r] *= cr;
        }

        // build P A-operand directly in registers (row owner has all values)
        v16bf pa;
        if (g == 0) {
#pragma unroll
            for (int j = 0; j < 8; ++j) {
                pa[j]     = (__bf16)sr[j];        // K = j
                pa[j + 8] = (__bf16)sr[j + 8];    // K = j+8
            }
        } else {
#pragma unroll
            for (int j = 0; j < 8; ++j) {
                pa[j]     = (__bf16)sr[j + 16];   // K = j+16
                pa[j + 8] = (__bf16)sr[j + 24];   // K = j+24
            }
        }

        // ---- O += P @ V over this 32-key chunk ----
#pragma unroll
        for (int ct = 0; ct < 8; ++ct) {
            v16bf vreg = vl[ct * 32 + lane];
            o[ct] = __builtin_amdgcn_wmma_f32_16x16x32_bf16(
                false, pa, false, vreg, (short)0, o[ct], false, false);
        }
        __syncthreads();  // all reads of buf `cur` done before it is refilled
    }

    // ---- normalize by row sums and store [b][c][n] ----
    if (lane < 16) sRow[w][row] = lrow;
    __syncthreads();
#pragma unroll
    for (int r = 0; r < 8; ++r) {
        float inv = 1.0f / sRow[w][r + 8 * g];
        int n = qt * 16 + r + 8 * g;
#pragma unroll
        for (int ct = 0; ct < 8; ++ct) {
            int c = ct * 16 + row;
            out[((size_t)b * 128 + c) * N + n] = o[ct][r] * inv;
        }
    }
}

// ---------------------------------------------------------------------------
// Half-pixel bilinear upsample (align_corners=False) of scale-2 and scale-4
// attention outputs, added onto the scale-1 result already in d_out.
// ---------------------------------------------------------------------------
__device__ __forceinline__ float bil(const float* __restrict__ src, int S,
                                     int h, int w) {
    float fy = (h + 0.5f) * (S * 0.015625f) - 0.5f;  // * S/64
    float fx = (w + 0.5f) * (S * 0.015625f) - 0.5f;
    int y0 = (int)floorf(fy), x0 = (int)floorf(fx);
    float ty = fy - y0, tx = fx - x0;
    int y1 = min(y0 + 1, S - 1), x1 = min(x0 + 1, S - 1);
    y0 = max(y0, 0); x0 = max(x0, 0);
    float v00 = src[y0 * S + x0], v01 = src[y0 * S + x1];
    float v10 = src[y1 * S + x0], v11 = src[y1 * S + x1];
    return v00 * (1.f - ty) * (1.f - tx) + v01 * (1.f - ty) * tx +
           v10 * ty * (1.f - tx) + v11 * ty * tx;
}

__global__ void upsample_add(const float* __restrict__ o2,
                             const float* __restrict__ o4,
                             float* __restrict__ out) {
    int idx = blockIdx.x * blockDim.x + threadIdx.x;  // B*C*64*64 = 2^21
    int w = idx & 63, h = (idx >> 6) & 63;
    int bc = idx >> 12;  // b*128 + c
    const float* p2 = o2 + (size_t)bc * 1024;  // 32x32 plane
    const float* p4 = o4 + (size_t)bc * 256;   // 16x16 plane
    out[idx] += bil(p2, 32, h, w) + bil(p4, 16, h, w);
}

// ---------------------------------------------------------------------------
extern "C" void kernel_launch(void* const* d_in, const int* in_sizes, int n_in,
                              void* d_out, int out_size, void* d_ws, size_t ws_size,
                              hipStream_t stream) {
    const float* x = (const float*)d_in[0];
    float* out = (float*)d_out;

    char* ws = (char*)d_ws;
    size_t off = 0;
    auto alloc = [&](size_t bytes) -> void* {
        void* p = ws + off;
        off += (bytes + 255) & ~(size_t)255;
        return p;
    };

    const int Ns[3] = {4096, 1024, 256};
    const int ss[3] = {1, 2, 4};

    __bf16* qa[3]; __bf16* kb[3]; __bf16* vb[3];
    for (int i = 0; i < 3; ++i) {
        size_t halves = (size_t)4 * Ns[i] * 128;  // B*N*C packed bf16 elems
        qa[i] = (__bf16*)alloc(halves * 2);
        kb[i] = (__bf16*)alloc(halves * 2);
        vb[i] = (__bf16*)alloc(halves * 2);
    }
    float* o2 = (float*)alloc((size_t)4 * 128 * 1024 * 4);
    float* o4 = (float*)alloc((size_t)4 * 128 * 256 * 4);
    float* outs[3] = {out, o2, o4};  // scale-1 writes straight into d_out

    for (int i = 0; i < 3; ++i) {
        int N = Ns[i], s = ss[i], Ws = 64 / s;
        int total = 4 * N * 128;
        int blocks = (total + 255) / 256;
        pack_qa<<<blocks, 256, 0, stream>>>(x, qa[i], N, s, Ws);
        pack_kb<<<blocks, 256, 0, stream>>>(x, kb[i], N, s, Ws);
        pack_vb<<<blocks, 256, 0, stream>>>(x, vb[i], N, s, Ws);
        flash_attn<<<dim3(N / 64, 4), 128, 0, stream>>>(qa[i], kb[i], vb[i],
                                                        outs[i], N);
    }
    upsample_add<<<(4 * 128 * 64 * 64) / 256, 256, 0, stream>>>(o2, o4, out);
}
